// MambaBottleneck_20358144983663
// MI455X (gfx1250) — matrix operand order repaired and checked by reading, not compiled, not run on hardware
//
#include <hip/hip_runtime.h>
#include <hip/hip_bf16.h>
#include <math.h>

// ---------------- problem constants ----------------
#define DIMC     128          // model dim (LayerNorm width, out channels)
#define DINNER   256          // expanded dim
#define NSTATE   16
#define DTRANK   8
#define NSEQ     8192         // L = D*H*W = 8*32*32
#define NBATCH   2
#define MTOT     (NBATCH*NSEQ)   // 16384 rows
#define XDLD     48           // padded leading dim for x_dbl (40 -> 48)
#define CSZ      128          // scan chunk length
#define NCHUNK   64           // NSEQ / CSZ

typedef _Float16 v16h __attribute__((ext_vector_type(16)));
typedef _Float16 v8h  __attribute__((ext_vector_type(8)));
typedef float    v8f  __attribute__((ext_vector_type(8)));
typedef float    v4f  __attribute__((ext_vector_type(4)));

__device__ __forceinline__ v8f mb_zero8() {
  v8f z;
#pragma unroll
  for (int i = 0; i < 8; ++i) z[i] = 0.0f;
  return z;
}

// A fragment: 16x32 f16, row-major source (lda halves per row).
// lane<16 : row = m0+lane,    halves[0..7]=K(k0..k0+7),   halves[8..15]=K(k0+16..k0+23)
// lane>=16: row = m0+lane-16, halves[0..7]=K(k0+8..k0+15),halves[8..15]=K(k0+24..k0+31)
__device__ __forceinline__ v16h mb_load_a(const _Float16* __restrict__ A, int lda,
                                          int m0, int k0, int lane) {
  int row = m0 + (lane & 15);
  int kb  = k0 + ((lane >> 4) ? 8 : 0);
  const _Float16* p = A + (size_t)row * lda + kb;
  v8h lo = *reinterpret_cast<const v8h*>(p);
  v8h hi = *reinterpret_cast<const v8h*>(p + 16);
  v16h r;
#pragma unroll
  for (int i = 0; i < 8; ++i) { r[i] = lo[i]; r[i + 8] = hi[i]; }
  return r;
}

// B fragment: 32x16 f16 (K x N). Source W is row-major (N rows, K cols): B[k][n]=W[n][k].
// lane<16 : col n = n0+lane,    halves[0..7]=K(k0..k0+7),   halves[8..15]=K(k0+16..k0+23)
// lane>=16: col n = n0+lane-16, halves[0..7]=K(k0+8..k0+15),halves[8..15]=K(k0+24..k0+31)
__device__ __forceinline__ v16h mb_load_b(const _Float16* __restrict__ W, int ldw,
                                          int n0, int k0, int lane) {
  int row = n0 + (lane & 15);
  int kb  = k0 + ((lane >> 4) ? 8 : 0);
  const _Float16* p = W + (size_t)row * ldw + kb;
  v8h lo = *reinterpret_cast<const v8h*>(p);
  v8h hi = *reinterpret_cast<const v8h*>(p + 16);
  v16h r;
#pragma unroll
  for (int i = 0; i < 8; ++i) { r[i] = lo[i]; r[i + 8] = hi[i]; }
  return r;
}

// ---------------- 1) LayerNorm over C=128, emit f16 activations ----------------
// x layout: (B, C, L) fp32.  xf16 layout: (MTOT, 128) row-major f16, m = b*L + l.
__global__ void mb_ln_kernel(const float* __restrict__ x, const float* __restrict__ g,
                             const float* __restrict__ beta, _Float16* __restrict__ xf16) {
  int wave = threadIdx.x >> 5;
  int lane = threadIdx.x & 31;
  int m = blockIdx.x * 8 + wave;
  int b = m >> 13;
  int l = m & (NSEQ - 1);
  const float* xp = x + (size_t)b * DIMC * NSEQ + l;
  float v[4], s = 0.0f, ss = 0.0f;
#pragma unroll
  for (int j = 0; j < 4; ++j) {
    int c = lane + 32 * j;
    v[j] = xp[(size_t)c * NSEQ];
    s += v[j]; ss += v[j] * v[j];
  }
#pragma unroll
  for (int mask = 16; mask >= 1; mask >>= 1) {
    s  += __shfl_xor(s,  mask, 32);
    ss += __shfl_xor(ss, mask, 32);
  }
  float mu  = s * (1.0f / DIMC);
  float var = ss * (1.0f / DIMC) - mu * mu;
  float inv = rsqrtf(var + 1e-5f);
#pragma unroll
  for (int j = 0; j < 4; ++j) {
    int c = lane + 32 * j;
    float nv = (v[j] - mu) * inv * g[c] + beta[c];
    xf16[(size_t)m * DIMC + c] = (_Float16)nv;
  }
}

// ---------------- 2) weight f32 -> f16 conversion (x_proj padded to 48 rows) ----
__global__ void mb_wcvt_kernel(const float* __restrict__ w1, const float* __restrict__ wx,
                               const float* __restrict__ wo, _Float16* __restrict__ w1h,
                               _Float16* __restrict__ wxh, _Float16* __restrict__ woh) {
  int id = blockIdx.x * 256 + threadIdx.x;
  if (id < 512 * 128) { w1h[id] = (_Float16)w1[id]; return; }
  id -= 512 * 128;
  if (id < XDLD * DINNER) {
    int row = id / DINNER, col = id % DINNER;
    wxh[id] = (row < DTRANK + 2 * NSTATE) ? (_Float16)wx[row * DINNER + col] : (_Float16)0.0f;
    return;
  }
  id -= XDLD * DINNER;
  if (id < DIMC * DINNER) woh[id] = (_Float16)wo[id];
}

// ---------------- 3) in_proj GEMM: (16384x128)x(128->512) via WMMA ----------------
__global__ void mb_inproj_kernel(const _Float16* __restrict__ A, const _Float16* __restrict__ Wh,
                                 float* __restrict__ xz) {
  int lane  = threadIdx.x & 31;
  int wave  = threadIdx.x >> 5;
  int mtile = blockIdx.y * 128 + wave * 16;
  int ntile = blockIdx.x * 64;
  v8f acc[4];
#pragma unroll
  for (int j = 0; j < 4; ++j) acc[j] = mb_zero8();
#pragma unroll
  for (int kk = 0; kk < 128; kk += 32) {
    v16h af = mb_load_a(A, 128, mtile, kk, lane);
#pragma unroll
    for (int j = 0; j < 4; ++j) {
      v16h bf = mb_load_b(Wh, 128, ntile + 16 * j, kk, lane);
      acc[j] = __builtin_amdgcn_wmma_f32_16x16x32_f16(false, af, false, bf,
                                                      (short)0, acc[j], false, false);
    }
  }
  int nloc = lane & 15;
  int mrow = mtile + (lane >> 4) * 8;
#pragma unroll
  for (int j = 0; j < 4; ++j) {
    int n = ntile + 16 * j + nloc;
#pragma unroll
    for (int r = 0; r < 8; ++r)
      xz[(size_t)(mrow + r) * 512 + n] = acc[j][r];
  }
}

// ---------------- 4) depthwise causal conv(4) + SiLU ----------------
__global__ void mb_conv_kernel(const float* __restrict__ xz, const float* __restrict__ cw,
                               const float* __restrict__ cb, float* __restrict__ uf,
                               _Float16* __restrict__ uh) {
  int id = blockIdx.x * 256 + threadIdx.x;
  int m = id >> 8;
  int d = id & 255;
  int l = m & (NSEQ - 1);
  int b = m >> 13;
  float acc = cb[d];
#pragma unroll
  for (int t = 0; t < 4; ++t) {
    int ll = l - 3 + t;
    if (ll >= 0) acc += xz[(size_t)((b << 13) + ll) * 512 + d] * cw[d * 4 + t];
  }
  float u = acc / (1.0f + __expf(-acc));
  uf[(size_t)m * DINNER + d] = u;
  uh[(size_t)m * DINNER + d] = (_Float16)u;
}

// ---------------- 5) x_proj GEMM: (16384x256)x(256->48pad) via WMMA ----------------
__global__ void mb_xproj_kernel(const _Float16* __restrict__ uh, const _Float16* __restrict__ Wxh,
                                float* __restrict__ xdbl) {
  int lane  = threadIdx.x & 31;
  int wave  = threadIdx.x >> 5;
  int mtile = blockIdx.y * 128 + wave * 16;
  int ntile = blockIdx.x * 16;
  v8f acc = mb_zero8();
#pragma unroll
  for (int kk = 0; kk < 256; kk += 32) {
    v16h af = mb_load_a(uh, 256, mtile, kk, lane);
    v16h bf = mb_load_b(Wxh, 256, ntile, kk, lane);
    acc = __builtin_amdgcn_wmma_f32_16x16x32_f16(false, af, false, bf,
                                                 (short)0, acc, false, false);
  }
  int n    = ntile + (lane & 15);
  int mrow = mtile + (lane >> 4) * 8;
#pragma unroll
  for (int r = 0; r < 8; ++r)
    xdbl[(size_t)(mrow + r) * XDLD + n] = acc[r];
}

// ---------------- 6) dt = softplus(dt_low @ dt_proj_w^T + b) ----------------
__global__ void mb_dt_kernel(const float* __restrict__ xdbl, const float* __restrict__ dw,
                             const float* __restrict__ db, float* __restrict__ dt) {
  int id = blockIdx.x * 256 + threadIdx.x;
  int m = id >> 8;
  int d = id & 255;
  float acc = db[d];
  const float* row = xdbl + (size_t)m * XDLD;
#pragma unroll
  for (int r = 0; r < DTRANK; ++r) acc += row[r] * dw[d * DTRANK + r];
  dt[(size_t)m * DINNER + d] = (acc > 20.0f) ? acc : log1pf(__expf(acc));
}

// ---------------- 7a) chunked scan pass A: per-chunk local state + decay product --
// id bits: n=[3:0], d=[11:4], chunk=[17:12], b=[18]. 524288 threads.
__global__ void mb_scan_partial(const float* __restrict__ dt, const float* __restrict__ uf,
                                const float* __restrict__ xdbl, const float* __restrict__ alog,
                                float* __restrict__ hpart, float* __restrict__ pprod) {
  int id = blockIdx.x * 256 + threadIdx.x;
  int n     = id & 15;
  int d     = (id >> 4) & 255;
  int chunk = (id >> 12) & (NCHUNK - 1);
  int b     = id >> 18;
  float Adn = -__expf(alog[d * NSTATE + n]);
  float h = 0.0f, P = 1.0f;
  size_t mbase = (size_t)b * NSEQ + (size_t)chunk * CSZ;
  for (int l = 0; l < CSZ; ++l) {
    size_t m = mbase + l;
    float dtv = dt[m * DINNER + d];
    float uv  = uf[m * DINNER + d];
    float Bv  = xdbl[m * XDLD + DTRANK + n];
    float dA  = __expf(dtv * Adn);
    h = dA * h + (dtv * uv) * Bv;
    P *= dA;
  }
  hpart[id] = h;
  pprod[id] = P;
}

// ---------------- 7b) chunked scan pass B: exclusive carry across chunks ----------
// 8192 threads, 64 serial combine steps each: carry_c = h_local_c + P_c * carry_{c-1}.
__global__ void mb_scan_carry(const float* __restrict__ hpart, const float* __restrict__ pprod,
                              float* __restrict__ carryin) {
  int id = blockIdx.x * 256 + threadIdx.x;   // (b, d, n)
  int n = id & 15;
  int d = (id >> 4) & 255;
  int b = id >> 12;
  float carry = 0.0f;
  for (int c = 0; c < NCHUNK; ++c) {
    int idx = ((b * NCHUNK + c) << 12) + (d << 4) + n;
    carryin[idx] = carry;
    carry = hpart[idx] + pprod[idx] * carry;
  }
}

// ---------------- 7c) chunked scan pass C: re-run chunk with carry-in, emit y -----
// block = 8 waves = 16 channels; lane = n + 16*(channel parity).
__global__ void mb_scan_final(const float* __restrict__ dt, const float* __restrict__ uf,
                              const float* __restrict__ xdbl, const float* __restrict__ alog,
                              const float* __restrict__ carryin, float* __restrict__ ys) {
  int lane  = threadIdx.x & 31;
  int wave  = threadIdx.x >> 5;
  int cg    = blockIdx.x & 15;                 // channel group of 16
  int chunk = (blockIdx.x >> 4) & (NCHUNK - 1);
  int b     = blockIdx.x >> 10;
  int d = cg * 16 + wave * 2 + (lane >> 4);
  int n = lane & 15;
  float Adn = -__expf(alog[d * NSTATE + n]);
  int idx = ((b * NCHUNK + chunk) << 12) + (d << 4) + n;
  float h = carryin[idx];
  size_t mbase = (size_t)b * NSEQ + (size_t)chunk * CSZ;
  for (int l = 0; l < CSZ; ++l) {
    size_t m = mbase + l;
    float dtv = dt[m * DINNER + d];
    float uv  = uf[m * DINNER + d];
    float Bv  = xdbl[m * XDLD + DTRANK + n];
    float Cv  = xdbl[m * XDLD + DTRANK + NSTATE + n];
    float dA  = __expf(dtv * Adn);
    h = dA * h + (dtv * uv) * Bv;
    float y = h * Cv;
    y += __shfl_xor(y, 1, 32);
    y += __shfl_xor(y, 2, 32);
    y += __shfl_xor(y, 4, 32);
    y += __shfl_xor(y, 8, 32);
    if (n == 0) ys[m * DINNER + d] = y;
  }
}

// ---------------- 8) gate: y = (ys + u*D) * silu(z), emit f16 ----------------
__global__ void mb_gate_kernel(const float* __restrict__ ys, const float* __restrict__ uf,
                               const float* __restrict__ Dp, const float* __restrict__ xz,
                               _Float16* __restrict__ yh) {
  int id = blockIdx.x * 256 + threadIdx.x;
  int m = id >> 8;
  int d = id & 255;
  float y = ys[(size_t)m * DINNER + d] + uf[(size_t)m * DINNER + d] * Dp[d];
  float z = xz[(size_t)m * 512 + DINNER + d];
  float sz = z / (1.0f + __expf(-z));
  yh[(size_t)m * DINNER + d] = (_Float16)(y * sz);
}

// ---------------- 9) out_proj GEMM + residual, scatter to (B,C,L) ----------------
__global__ void mb_outproj_kernel(const _Float16* __restrict__ yh, const _Float16* __restrict__ Woh,
                                  const float* __restrict__ x, float* __restrict__ out) {
  int lane  = threadIdx.x & 31;
  int wave  = threadIdx.x >> 5;
  int mtile = blockIdx.y * 128 + wave * 16;
  int ntile = blockIdx.x * 64;
  v8f acc[4];
#pragma unroll
  for (int j = 0; j < 4; ++j) acc[j] = mb_zero8();
#pragma unroll
  for (int kk = 0; kk < 256; kk += 32) {
    v16h af = mb_load_a(yh, 256, mtile, kk, lane);
#pragma unroll
    for (int j = 0; j < 4; ++j) {
      v16h bf = mb_load_b(Woh, 256, ntile + 16 * j, kk, lane);
      acc[j] = __builtin_amdgcn_wmma_f32_16x16x32_f16(false, af, false, bf,
                                                      (short)0, acc[j], false, false);
    }
  }
  int b     = mtile >> 13;
  int lbase = (mtile & (NSEQ - 1)) + (lane >> 4) * 8;
  int nloc  = lane & 15;
#pragma unroll
  for (int j = 0; j < 4; ++j) {
    int c = ntile + 16 * j + nloc;
    size_t base = (size_t)b * DIMC * NSEQ + (size_t)c * NSEQ + lbase;
    v4f r0 = *reinterpret_cast<const v4f*>(x + base);
    v4f r1 = *reinterpret_cast<const v4f*>(x + base + 4);
    v4f o0, o1;
#pragma unroll
    for (int i = 0; i < 4; ++i) { o0[i] = acc[j][i] + r0[i]; o1[i] = acc[j][i + 4] + r1[i]; }
    *reinterpret_cast<v4f*>(out + base)     = o0;
    *reinterpret_cast<v4f*>(out + base + 4) = o1;
  }
}

// ---------------- host launcher ----------------
extern "C" void kernel_launch(void* const* d_in, const int* in_sizes, int n_in,
                              void* d_out, int out_size, void* d_ws, size_t ws_size,
                              hipStream_t stream) {
  (void)in_sizes; (void)n_in; (void)out_size; (void)ws_size;
  const float* x    = (const float*)d_in[0];
  const float* ng   = (const float*)d_in[1];
  const float* nb   = (const float*)d_in[2];
  const float* w1   = (const float*)d_in[3];
  const float* cw   = (const float*)d_in[4];
  const float* cb   = (const float*)d_in[5];
  const float* wx   = (const float*)d_in[6];
  const float* dw   = (const float*)d_in[7];
  const float* db   = (const float*)d_in[8];
  const float* alog = (const float*)d_in[9];
  const float* Dp   = (const float*)d_in[10];
  const float* wo   = (const float*)d_in[11];
  float* out = (float*)d_out;

  char* ws = (char*)d_ws;
  size_t off = 0;
  auto carve = [&](size_t bytes) -> char* {
    char* p = ws + off;
    off += (bytes + 255) & ~(size_t)255;
    return p;
  };
  const size_t scan_elems = (size_t)NBATCH * NCHUNK * DINNER * NSTATE;  // 524288
  _Float16* xf16 = (_Float16*)carve((size_t)MTOT * DIMC * 2);
  _Float16* w1h  = (_Float16*)carve((size_t)512 * 128 * 2);
  _Float16* wxh  = (_Float16*)carve((size_t)XDLD * DINNER * 2);
  _Float16* woh  = (_Float16*)carve((size_t)DIMC * DINNER * 2);
  float*    xz   = (float*)   carve((size_t)MTOT * 512 * 4);
  float*    uf   = (float*)   carve((size_t)MTOT * DINNER * 4);
  _Float16* uh   = (_Float16*)carve((size_t)MTOT * DINNER * 2);
  float*    xdbl = (float*)   carve((size_t)MTOT * XDLD * 4);
  float*    dtb  = (float*)   carve((size_t)MTOT * DINNER * 4);
  float*    ysb  = (float*)   carve((size_t)MTOT * DINNER * 4);
  _Float16* yh   = (_Float16*)carve((size_t)MTOT * DINNER * 2);
  float*    hp   = (float*)   carve(scan_elems * 4);
  float*    pp   = (float*)   carve(scan_elems * 4);
  float*    ci   = (float*)   carve(scan_elems * 4);

  mb_ln_kernel<<<MTOT / 8, 256, 0, stream>>>(x, ng, nb, xf16);
  int wcvt_elems = 512 * 128 + XDLD * DINNER + DIMC * DINNER;
  mb_wcvt_kernel<<<(wcvt_elems + 255) / 256, 256, 0, stream>>>(w1, wx, wo, w1h, wxh, woh);
  mb_inproj_kernel<<<dim3(8, 128), 256, 0, stream>>>(xf16, w1h, xz);
  mb_conv_kernel<<<(MTOT * DINNER) / 256, 256, 0, stream>>>(xz, cw, cb, uf, uh);
  mb_xproj_kernel<<<dim3(3, 128), 256, 0, stream>>>(uh, wxh, xdbl);
  mb_dt_kernel<<<(MTOT * DINNER) / 256, 256, 0, stream>>>(xdbl, dw, db, dtb);
  mb_scan_partial<<<(int)(scan_elems / 256), 256, 0, stream>>>(dtb, uf, xdbl, alog, hp, pp);
  mb_scan_carry<<<(NBATCH * DINNER * NSTATE) / 256, 256, 0, stream>>>(hp, pp, ci);
  mb_scan_final<<<NBATCH * NCHUNK * (DINNER / 16), 256, 0, stream>>>(dtb, uf, xdbl, alog, ci, ysb);
  mb_gate_kernel<<<(MTOT * DINNER) / 256, 256, 0, stream>>>(ysb, uf, Dp, xz, yh);
  mb_outproj_kernel<<<dim3(2, 128), 256, 0, stream>>>(yh, woh, x, out);
}
